// InfluencerRankModel_42640435315011
// MI455X (gfx1250) — matrix operand order repaired and checked by reading, not compiled, not run on hardware
//
#include <hip/hip_runtime.h>
#include <hip/hip_bf16.h>

typedef __attribute__((ext_vector_type(16))) _Float16 v16h;
typedef __attribute__((ext_vector_type(8)))  _Float16 v8h;
typedef __attribute__((ext_vector_type(8)))  float    v8f;

#define BM 128
#define BN 128
#define BK 64
#define BKP 72   // padded LDS row stride (f16 elems) -> 144B, conflict-free fragment reads

// ---------------------------------------------------------------------------
// CDNA5 async-copy helpers (memory -> LDS, tracked by ASYNCcnt)
// ---------------------------------------------------------------------------
__device__ __forceinline__ unsigned lds_off_u32(const void* p) {
    // generic pointer to __shared__ has the LDS byte offset in its low 32 bits
    return (unsigned)(size_t)p;
}
__device__ __forceinline__ void async_copy_b128(unsigned lds_off, const void* gaddr) {
    unsigned long long ga = (unsigned long long)(size_t)gaddr;
    asm volatile("global_load_async_to_lds_b128 %0, %1, off"
                 :: "v"(lds_off), "v"(ga) : "memory");
}
__device__ __forceinline__ void wait_async0() {
    asm volatile("s_wait_asynccnt 0x0" ::: "memory");
}

// ---------------------------------------------------------------------------
// WMMA GEMM: C[M x Nc] = A[M x K] * B, with B given as BT[Nc x K] (f16 row-major)
// A is f16 row-major with row stride lda (elements). C is f32 row-major.
// Double-buffered LDS staging via global_load_async_to_lds_b128.
// ---------------------------------------------------------------------------
__global__ __launch_bounds__(256)
void gemm_f16_wmma(const _Float16* __restrict__ A, int lda,
                   const _Float16* __restrict__ BT,
                   float* __restrict__ C,
                   int M, int K, int Nc)
{
    __shared__ _Float16 As[2][BM * BKP];
    __shared__ _Float16 Bs[2][BN * BKP];

    const int tid  = threadIdx.x;
    const int lane = tid & 31;
    const int wid  = tid >> 5;
    const int wr   = wid >> 1;   // 0..3 : 32-row stripe
    const int wc   = wid & 1;    // 0..1 : 64-col stripe
    const int lr   = lane & 15;
    const int hi   = lane >> 4;  // 0 or 1

    const int m0 = blockIdx.x * BM;
    const int n0 = blockIdx.y * BN;

    v8f acc[2][4];
#pragma unroll
    for (int i = 0; i < 2; i++)
#pragma unroll
        for (int j = 0; j < 4; j++)
#pragma unroll
            for (int r = 0; r < 8; r++) acc[i][j][r] = 0.0f;

    // Edge block: zero OOB A rows once in both buffers; async stages skip those
    // lanes, so the zeros persist across all k-tiles.
    if (m0 + BM > M) {
        for (int i = tid; i < BM * BKP; i += 256) {
            As[0][i] = (_Float16)0.0f;
            As[1][i] = (_Float16)0.0f;
        }
    }

    // issue async stage of one (BM x BK) A tile + (BN x BK) B tile into buffer `buf`
    auto stage = [&](int buf, int k0) {
        _Float16* Ab = &As[buf][0];
        _Float16* Bb = &Bs[buf][0];
#pragma unroll
        for (int i = 0; i < 4; i++) {
            int s    = tid + i * 256;      // 0..1023
            int row  = s >> 3;             // 0..127
            int cseg = (s & 7) * 8;        // 0..56
            int gm   = m0 + row;
            if (gm < M)
                async_copy_b128(lds_off_u32(Ab + row * BKP + cseg),
                                A + (size_t)gm * lda + k0 + cseg);
        }
#pragma unroll
        for (int i = 0; i < 4; i++) {
            int s    = tid + i * 256;
            int row  = s >> 3;
            int cseg = (s & 7) * 8;
            async_copy_b128(lds_off_u32(Bb + row * BKP + cseg),
                            BT + (size_t)(n0 + row) * K + k0 + cseg);
        }
    };

    stage(0, 0);
    wait_async0();
    __syncthreads();

    const int ntiles = K / BK;
    for (int kt = 0; kt < ntiles; kt++) {
        const int cur = kt & 1;
        if (kt + 1 < ntiles) stage(cur ^ 1, (kt + 1) * BK);

        const _Float16* Ac = &As[cur][0];
        const _Float16* Bc = &Bs[cur][0];
#pragma unroll
        for (int ks = 0; ks < 2; ks++) {
            const int kb = ks * 32;
            v16h af[2], bf[4];
            // A fragment: lane<16 holds K [kb..kb+7],[kb+16..23]; lane>=16 the +8 halves
#pragma unroll
            for (int mt = 0; mt < 2; mt++) {
                int r = wr * 32 + mt * 16 + lr;
                v8h lo = *(const v8h*)(Ac + r * BKP + kb + hi * 8);
                v8h hh = *(const v8h*)(Ac + r * BKP + kb + 16 + hi * 8);
                af[mt] = __builtin_shufflevector(lo, hh, 0,1,2,3,4,5,6,7,8,9,10,11,12,13,14,15);
            }
            // B fragment: lane n holds column n, K [kb..kb+7],[kb+16..23] (lane>=16: the +8 halves)
#pragma unroll
            for (int nt = 0; nt < 4; nt++) {
                int c = wc * 64 + nt * 16 + lr;
                v8h lo = *(const v8h*)(Bc + c * BKP + kb + hi * 8);
                v8h hh = *(const v8h*)(Bc + c * BKP + kb + 16 + hi * 8);
                bf[nt] = __builtin_shufflevector(lo, hh, 0,1,2,3,4,5,6,7,8,9,10,11,12,13,14,15);
            }
#pragma unroll
            for (int mt = 0; mt < 2; mt++)
#pragma unroll
                for (int nt = 0; nt < 4; nt++)
                    acc[mt][nt] = __builtin_amdgcn_wmma_f32_16x16x32_f16(
                        false, af[mt], false, bf[nt], (short)0, acc[mt][nt], false, false);
        }

        wait_async0();       // next tile resident in LDS (own wave's copies)
        __syncthreads();     // all waves' copies visible; everyone done reading `cur`
    }

    // ---- store C: VGPR r -> row (hi*8 + r), lane lr -> column ----
#pragma unroll
    for (int mt = 0; mt < 2; mt++) {
        int rbase = m0 + wr * 32 + mt * 16 + hi * 8;
#pragma unroll
        for (int nt = 0; nt < 4; nt++) {
            int col = n0 + wc * 64 + nt * 16 + lr;
#pragma unroll
            for (int r = 0; r < 8; r++) {
                int row = rbase + r;
                if (row < M) C[(size_t)row * Nc + col] = acc[mt][nt][r];
            }
        }
    }
}

// ---------------------------------------------------------------------------
// Elementwise / graph kernels
// ---------------------------------------------------------------------------
__global__ void zero_f32_kernel(float* p, long long n) {
    long long i = (long long)blockIdx.x * blockDim.x + threadIdx.x;
    if (i < n) p[i] = 0.0f;
}
__global__ void zero_f16_kernel(_Float16* p, long long n) {
    long long i = (long long)blockIdx.x * blockDim.x + threadIdx.x;
    if (i < n) p[i] = (_Float16)0.0f;
}
__global__ void f32_to_f16_kernel(const float* __restrict__ s, _Float16* __restrict__ d, long long n) {
    long long i = (long long)blockIdx.x * blockDim.x + threadIdx.x;
    if (i < n) d[i] = (_Float16)s[i];
}
// WT[n*K + k] = W[k*Nc + n]  (build BT layout for GCN weights)
__global__ void transpose_f32_to_f16_kernel(const float* __restrict__ W, _Float16* __restrict__ WT,
                                            int K, int Nc) {
    int idx = blockIdx.x * blockDim.x + threadIdx.x;
    if (idx >= K * Nc) return;
    int n = idx / K, k = idx - n * K;
    WT[idx] = (_Float16)W[k * Nc + n];
}
__global__ void deg_kernel(const int* __restrict__ dst, float* __restrict__ deg, int E) {
    int e = blockIdx.x * blockDim.x + threadIdx.x;
    if (e < E) atomicAdd(&deg[dst[e]], 1.0f);
}
__global__ void dinv_kernel(float* d, int N) {
    int i = blockIdx.x * blockDim.x + threadIdx.x;
    if (i < N) d[i] = rsqrtf(d[i] + 1.0f);   // +1 self loop
}
__global__ void norm_kernel(const int* __restrict__ src, const int* __restrict__ dst,
                            const float* __restrict__ dinv, float* __restrict__ ne, int E) {
    int e = blockIdx.x * blockDim.x + threadIdx.x;
    if (e < E) ne[e] = dinv[src[e]] * dinv[dst[e]];
}
// agg = self_norm * hw + bias
__global__ void agg_init_kernel(const float* __restrict__ hw, const float* __restrict__ dinv,
                                const float* __restrict__ b, float* __restrict__ agg, int N) {
    int idx = blockIdx.x * blockDim.x + threadIdx.x;
    if (idx >= N * 256) return;
    int i = idx >> 8, k = idx & 255;
    float dv = dinv[i];
    agg[idx] = dv * dv * hw[idx] + b[k];
}
// 64 threads per edge, float4 per thread: agg[dst] += norm_e * hw[src]
__global__ void scatter_kernel(const int* __restrict__ src, const int* __restrict__ dst,
                               const float* __restrict__ ne, const float* __restrict__ hw,
                               float* __restrict__ agg, int E) {
    long long idx = (long long)blockIdx.x * blockDim.x + threadIdx.x;
    int e = (int)(idx >> 6);
    if (e >= E) return;
    int q = ((int)idx & 63) * 4;
    float c = ne[e];
    const float4 v = *(const float4*)(hw + (size_t)src[e] * 256 + q);
    float* a = agg + (size_t)dst[e] * 256 + q;
    atomicAdd(a + 0, c * v.x);
    atomicAdd(a + 1, c * v.y);
    atomicAdd(a + 2, c * v.z);
    atomicAdd(a + 3, c * v.w);
}
// relu(agg) -> f16 into dst with row stride (writes one 256-wide slice of emb)
__global__ void relu_store_kernel(const float* __restrict__ agg, _Float16* __restrict__ dst,
                                  int stride, int N) {
    int idx = blockIdx.x * blockDim.x + threadIdx.x;
    if (idx >= N * 256) return;
    int i = idx >> 8, k = idx & 255;
    dst[(size_t)i * stride + k] = (_Float16)fmaxf(agg[idx], 0.0f);
}
// GRU gate fusion: consumes gi [N,768], gh [N,768]; updates h, h16, hs[t]
__global__ void gate_kernel(const float* __restrict__ gi, const float* __restrict__ gh,
                            const float* __restrict__ bih, const float* __restrict__ bhh,
                            float* __restrict__ h, _Float16* __restrict__ h16,
                            _Float16* __restrict__ hs_t, int N) {
    int idx = blockIdx.x * blockDim.x + threadIdx.x;
    if (idx >= N * 256) return;
    int i = idx >> 8, j = idx & 255;
    size_t b = (size_t)i * 768;
    float ir = gi[b + j]       + bih[j];
    float iz = gi[b + 256 + j] + bih[256 + j];
    float in = gi[b + 512 + j] + bih[512 + j];
    float hr = gh[b + j]       + bhh[j];
    float hz = gh[b + 256 + j] + bhh[256 + j];
    float hn = gh[b + 512 + j] + bhh[512 + j];
    float r = 1.0f / (1.0f + __expf(-(ir + hr)));
    float z = 1.0f / (1.0f + __expf(-(iz + hz)));
    float n = tanhf(in + r * hn);
    float hv = (1.0f - z) * n + z * h[idx];
    h[idx]   = hv;
    h16[idx] = (_Float16)hv;
    hs_t[idx] = (_Float16)hv;
}
// scores[item] = tanh(dot(hs[item], Wa) + ba); one wave per (t,i)
__global__ void score_kernel(const _Float16* __restrict__ hs, const float* __restrict__ Wa,
                             const float* __restrict__ ba, float* __restrict__ sc, int TN) {
    int item = blockIdx.x * 8 + (threadIdx.x >> 5);
    int lane = threadIdx.x & 31;
    if (item >= TN) return;
    const _Float16* row = hs + (size_t)item * 256;
    float s = 0.0f;
#pragma unroll
    for (int k = lane; k < 256; k += 32) s += (float)row[k] * Wa[k];
#pragma unroll
    for (int o = 16; o > 0; o >>= 1) s += __shfl_xor(s, o, 32);
    if (lane == 0) sc[item] = tanhf(s + ba[0]);
}
__global__ void softmax_time_kernel(float* __restrict__ w, int N, int T) {
    int i = blockIdx.x * blockDim.x + threadIdx.x;
    if (i >= N) return;
    float v[12], m = -1e30f;
    for (int t = 0; t < T; t++) { v[t] = w[t * N + i]; m = fmaxf(m, v[t]); }
    float s = 0.0f;
    for (int t = 0; t < T; t++) { v[t] = __expf(v[t] - m); s += v[t]; }
    float inv = 1.0f / s;
    for (int t = 0; t < T; t++) w[t * N + i] = v[t] * inv;
}
__global__ void ctx_kernel(const _Float16* __restrict__ hs, const float* __restrict__ w,
                           float* __restrict__ ctx, int N, int T) {
    int idx = blockIdx.x * blockDim.x + threadIdx.x;
    if (idx >= N * 256) return;
    int i = idx >> 8, k = idx & 255;
    float c = 0.0f;
    for (int t = 0; t < T; t++)
        c += (float)hs[((size_t)t * N + i) * 256 + k] * w[t * N + i];
    ctx[idx] = c;
}
// out[node] = relu(ctx @ Wp1 + bp1) @ Wp2 + bp2 ; 16 threads per node
__global__ __launch_bounds__(256)
void predictor_kernel(const float* __restrict__ ctx, const float* __restrict__ Wp1,
                      const float* __restrict__ bp1, const float* __restrict__ Wp2,
                      const float* __restrict__ bp2, float* __restrict__ out, int N) {
    __shared__ float Ws[256 * 16];
    __shared__ float bs[16];
    for (int i = threadIdx.x; i < 256 * 16; i += 256) Ws[i] = Wp1[i];
    if (threadIdx.x < 16) bs[threadIdx.x] = bp1[threadIdx.x];
    __syncthreads();
    int node = blockIdx.x * 16 + (threadIdx.x >> 4);
    int j = threadIdx.x & 15;
    float acc = bs[j];
    if (node < N) {
        const float* c = ctx + (size_t)node * 256;
        for (int k = 0; k < 256; k++) acc += c[k] * Ws[k * 16 + j];
    }
    float v = fmaxf(acc, 0.0f) * Wp2[j];
#pragma unroll
    for (int o = 8; o > 0; o >>= 1) v += __shfl_xor(v, o, 16);
    if (node < N && j == 0) out[node] = v + bp2[0];
}

// ---------------------------------------------------------------------------
extern "C" void kernel_launch(void* const* d_in, const int* in_sizes, int n_in,
                              void* d_out, int out_size, void* d_ws, size_t ws_size,
                              hipStream_t stream) {
    const float* x   = (const float*)d_in[0];
    const int*  eix  = (const int*)d_in[1];
    const float* Wg1 = (const float*)d_in[2];
    const float* bg1 = (const float*)d_in[3];
    const float* Wg2 = (const float*)d_in[4];
    const float* bg2 = (const float*)d_in[5];
    const float* Wih = (const float*)d_in[6];
    const float* Whh = (const float*)d_in[7];
    const float* bih = (const float*)d_in[8];
    const float* bhh = (const float*)d_in[9];
    const float* Wa  = (const float*)d_in[10];
    const float* ba  = (const float*)d_in[11];
    const float* Wp1 = (const float*)d_in[12];
    const float* bp1 = (const float*)d_in[13];
    const float* Wp2 = (const float*)d_in[14];
    const float* bp2 = (const float*)d_in[15];
    float* out = (float*)d_out;

    constexpr int T = 12, N = 20000, E = 640000;
    (void)in_sizes; (void)n_in; (void)out_size; (void)ws_size;

    char* ws = (char*)d_ws;
    size_t off = 0;
    auto alloc = [&](size_t bytes) -> char* {
        char* p = ws + off;
        off = (off + bytes + 255) & ~(size_t)255;
        return p;
    };
    float*    dinv  = (float*)alloc((size_t)N * 4);
    float*    nrm   = (float*)alloc((size_t)E * 4);
    _Float16* xh    = (_Float16*)alloc((size_t)N * 256 * 2);
    float*    hw    = (float*)alloc((size_t)N * 256 * 4);
    float*    agg   = (float*)alloc((size_t)N * 256 * 4);
    _Float16* emb   = (_Float16*)alloc((size_t)N * 512 * 2);
    float*    hbuf  = (float*)alloc((size_t)N * 256 * 4);
    _Float16* h16   = (_Float16*)alloc((size_t)N * 256 * 2);
    float*    gi    = (float*)alloc((size_t)N * 768 * 4);
    float*    gh    = (float*)alloc((size_t)N * 768 * 4);
    _Float16* hs    = (_Float16*)alloc((size_t)T * N * 256 * 2);
    float*    wt    = (float*)alloc((size_t)T * N * 4);
    float*    ctx   = (float*)alloc((size_t)N * 256 * 4);
    _Float16* Wg1T  = (_Float16*)alloc((size_t)256 * 256 * 2);
    _Float16* Wg2T  = (_Float16*)alloc((size_t)256 * 256 * 2);
    _Float16* Wih16 = (_Float16*)alloc((size_t)768 * 512 * 2);
    _Float16* Whh16 = (_Float16*)alloc((size_t)768 * 256 * 2);

    const int NE  = N * 256;              // 5.12M
    const int gNE = (NE + 255) / 256;
    const int gN  = (N + 255) / 256;
    const int gE  = (E + 255) / 256;

    // per-call weight prep (deterministic)
    transpose_f32_to_f16_kernel<<<(256 * 256 + 255) / 256, 256, 0, stream>>>(Wg1, Wg1T, 256, 256);
    transpose_f32_to_f16_kernel<<<(256 * 256 + 255) / 256, 256, 0, stream>>>(Wg2, Wg2T, 256, 256);
    f32_to_f16_kernel<<<(768 * 512 + 255) / 256, 256, 0, stream>>>(Wih, Wih16, 768LL * 512);
    f32_to_f16_kernel<<<(768 * 256 + 255) / 256, 256, 0, stream>>>(Whh, Whh16, 768LL * 256);
    zero_f32_kernel<<<gNE, 256, 0, stream>>>(hbuf, (long long)NE);
    zero_f16_kernel<<<gNE, 256, 0, stream>>>(h16, (long long)NE);

    for (int t = 0; t < T; t++) {
        const int* src = eix + (size_t)t * 2 * E;
        const int* dst = src + E;
        // degrees -> dinv -> edge norms
        zero_f32_kernel<<<gN, 256, 0, stream>>>(dinv, (long long)N);
        deg_kernel<<<gE, 256, 0, stream>>>(dst, dinv, E);
        dinv_kernel<<<gN, 256, 0, stream>>>(dinv, N);
        norm_kernel<<<gE, 256, 0, stream>>>(src, dst, dinv, nrm, E);
        // x[t] -> f16
        f32_to_f16_kernel<<<gNE, 256, 0, stream>>>(x + (size_t)t * NE, xh, (long long)NE);

        dim3 g1((N + BM - 1) / BM, 256 / BN);  // 157 x 2
        // GCN layer 1
        gemm_f16_wmma<<<g1, 256, 0, stream>>>(xh, 256, Wg1T, hw, N, 256, 256);
        agg_init_kernel<<<gNE, 256, 0, stream>>>(hw, dinv, bg1, agg, N);
        scatter_kernel<<<(E * 64) / 256, 256, 0, stream>>>(src, dst, nrm, hw, agg, E);
        relu_store_kernel<<<gNE, 256, 0, stream>>>(agg, emb, 512, N);       // h1 -> emb[:, :256]
        // GCN layer 2
        gemm_f16_wmma<<<g1, 256, 0, stream>>>(emb, 512, Wg2T, hw, N, 256, 256);
        agg_init_kernel<<<gNE, 256, 0, stream>>>(hw, dinv, bg2, agg, N);
        scatter_kernel<<<(E * 64) / 256, 256, 0, stream>>>(src, dst, nrm, hw, agg, E);
        relu_store_kernel<<<gNE, 256, 0, stream>>>(agg, emb + 256, 512, N); // h2 -> emb[:, 256:]

        // GRU step: gi = emb @ Wih^T ; gh = h @ Whh^T ; gates
        dim3 g2((N + BM - 1) / BM, 768 / BN);  // 157 x 6
        gemm_f16_wmma<<<g2, 256, 0, stream>>>(emb, 512, Wih16, gi, N, 512, 768);
        gemm_f16_wmma<<<g2, 256, 0, stream>>>(h16, 256, Whh16, gh, N, 256, 768);
        gate_kernel<<<gNE, 256, 0, stream>>>(gi, gh, bih, bhh, hbuf, h16,
                                             hs + (size_t)t * NE, N);
    }

    // attention over time + predictor
    score_kernel<<<(T * N + 7) / 8, 256, 0, stream>>>(hs, Wa, ba, wt, T * N);
    softmax_time_kernel<<<gN, 256, 0, stream>>>(wt, N, T);
    ctx_kernel<<<gNE, 256, 0, stream>>>(hs, wt, ctx, N, T);
    predictor_kernel<<<(N + 15) / 16, 256, 0, stream>>>(ctx, Wp1, bp1, Wp2, bp2, out, N);
}
